// PipeLine_9895604650538
// MI455X (gfx1250) — compile-verified, matrix operand order checked by reading, not run on hardware
//
#include <hip/hip_runtime.h>
#include <hip/hip_bf16.h>
#include <math.h>

typedef __attribute__((ext_vector_type(16))) _Float16 v16h;
typedef __attribute__((ext_vector_type(8)))  _Float16 v8h;
typedef __attribute__((ext_vector_type(8)))  float    v8f;

#define HW512 (512*512)

__device__ __forceinline__ float lrelu02(float v){ return v > 0.f ? v : 0.2f*v; }

// CDNA5 async global->LDS copy (ASYNCcnt-tracked, no VGPR round trip).
// vdst = VGPR holding LDS byte offset, vaddr = 64-bit global address.
__device__ __forceinline__ void async_copy_b128(unsigned lds_off, const _Float16* g) {
  asm volatile("global_load_async_to_lds_b128 %0, %1, off"
               :: "v"(lds_off), "v"(g)
               : "memory");
}
__device__ __forceinline__ void async_wait0() {
  asm volatile("s_wait_asynccnt 0x0" ::: "memory");
}

__device__ __forceinline__ void bilin_setup(float u, float v, int Wt, int Ht,
                                            int& x0, int& y0, float& wx, float& wy){
  float px = u * (float)(Wt - 1);
  float py = v * (float)(Ht - 1);
  float fx = fminf(fmaxf(floorf(px), 0.f), (float)(Wt - 2));
  float fy = fminf(fmaxf(floorf(py), 0.f), (float)(Ht - 2));
  x0 = (int)fx; y0 = (int)fy;
  wx = px - fx; wy = py - fy;
}

// ---------------------------------------------------------------------------
// Stage 1: albedo bilinear sample, S-reduction (fwd_term, f), nt features.
// nt written NHWC, 32-channel padded (zeros in 16..31).
// ---------------------------------------------------------------------------
__global__ __launch_bounds__(256) void stage1_kernel(
    const float* __restrict__ wi, const float* __restrict__ envmap,
    const float* __restrict__ uv_map, const float* __restrict__ extrinsics,
    const float* __restrict__ albedo_layer,
    const float* __restrict__ t1, const float* __restrict__ t2,
    const float* __restrict__ t3, const float* __restrict__ t4,
    float* __restrict__ out_fwd, float* __restrict__ out_albedo,
    float* __restrict__ f_buf, _Float16* __restrict__ nt_buf)
{
  const int H = 512, W = 512, S = 10;
  int gid = blockIdx.x * blockDim.x + threadIdx.x;
  if (gid >= 2 * H * W) return;
  int n = gid / (H * W);
  int p = gid - n * (H * W);
  int y = p / W, x = p - y * W;

  float u = uv_map[(size_t)((n*H + y)*W + x)*2 + 0];
  float v = uv_map[(size_t)((n*H + y)*W + x)*2 + 1];

  int ax0, ay0; float awx, awy;
  bilin_setup(u, v, 512, 512, ax0, ay0, awx, awy);
  float aw00 = (1.f-awx)*(1.f-awy), aw01 = awx*(1.f-awy);
  float aw10 = (1.f-awx)*awy,       aw11 = awx*awy;
  float alb[3];
  #pragma unroll
  for (int c = 0; c < 3; ++c) {
    const float* tp = albedo_layer + (size_t)c * HW512;
    alb[c] = tp[ay0*512+ax0]*aw00 + tp[ay0*512+ax0+1]*aw01
           + tp[(ay0+1)*512+ax0]*aw10 + tp[(ay0+1)*512+ax0+1]*aw11;
  }

  const float K0 = 0.28209479177387814f;
  const float K1 = 0.4886025119029199f;
  const float K2 = 1.0925484305920792f;
  const float K20 = 0.31539156525252005f;
  const float K22 = 0.5462742152960396f;
  float ts[9];
  #pragma unroll
  for (int k = 0; k < 9; ++k) ts[k] = 0.f;
  float es[3] = {0.f, 0.f, 0.f};
  for (int s = 0; s < S; ++s) {
    float th = wi[(((size_t)n*2 + 0)*S + s)*HW512 + p];
    float ph = wi[(((size_t)n*2 + 1)*S + s)*HW512 + p];
    float st = sinf(th), ct = cosf(th);
    float sp = sinf(ph), cp = cosf(ph);
    float xd = st*cp, yd = st*sp, zd = ct;
    ts[0] += K0;
    ts[1] += K1 * yd;
    ts[2] += K1 * zd;
    ts[3] += K1 * xd;
    ts[4] += K2 * xd * yd;
    ts[5] += K2 * yd * zd;
    ts[6] += K20 * (3.f*zd*zd - 1.f);
    ts[7] += K2 * xd * zd;
    ts[8] += K22 * (xd*xd - yd*yd);
    #pragma unroll
    for (int c = 0; c < 3; ++c)
      es[c] += envmap[(((size_t)n*3 + c)*S + s)*HW512 + p];
  }
  const float invS = 1.f / (float)S;
  #pragma unroll
  for (int c = 0; c < 3; ++c) {
    out_albedo[((size_t)n*3 + c)*HW512 + p] = alb[c];
    out_fwd   [((size_t)n*3 + c)*HW512 + p] = alb[c] * es[c] * invS;
  }
  #pragma unroll
  for (int k = 0; k < 9; ++k)
    #pragma unroll
    for (int c = 0; c < 3; ++c)
      f_buf[((size_t)n*27 + k*3 + c)*HW512 + p] = alb[c] * ts[k] * invS;

  float nt[16];
  #pragma unroll
  for (int c = 0; c < 16; ++c) nt[c] = 0.f;
  const float* texs[4] = { t1, t2, t3, t4 };
  const int    szs[4]  = { 512, 256, 128, 64 };
  for (int L = 0; L < 4; ++L) {
    int WT = szs[L];
    int tx0, ty0; float twx, twy;
    bilin_setup(u, v, WT, WT, tx0, ty0, twx, twy);
    float w00 = (1.f-twx)*(1.f-twy), w01 = twx*(1.f-twy);
    float w10 = (1.f-twx)*twy,       w11 = twx*twy;
    const float* base = texs[L];
    size_t hw = (size_t)WT * WT;
    for (int c = 0; c < 16; ++c) {
      const float* tp = base + (size_t)c * hw;
      nt[c] += tp[ty0*WT+tx0]*w00 + tp[ty0*WT+tx0+1]*w01
             + tp[(ty0+1)*WT+tx0]*w10 + tp[(ty0+1)*WT+tx0+1]*w11;
    }
  }
  {
    float ex = extrinsics[n*3+0], ey = extrinsics[n*3+1], ez = extrinsics[n*3+2];
    const float c0 = 0.28209479177387814f;
    const float c1 = 0.48860251190291987f;
    const float c2 = 1.0925484305920792f;
    const float c3 = 0.31539156525252005f;
    const float sq = 0.7071067811865476f;
    float bs[9];
    bs[0] = c0;
    bs[1] = ey * c1 * sq;
    bs[2] = ez * c1;
    bs[3] = -ex * c1 * sq;
    bs[4] = ex * ey * c2 * 0.5f * sq;
    bs[5] = ey * ez * c2 * sq;
    bs[6] = (3.f*ez*ez - 1.f) * c3;
    bs[7] = -ez * ex * c2 * sq;
    bs[8] = (ex*ex - ez*ez) * c2 * 0.5f * sq;
    #pragma unroll
    for (int j = 0; j < 9; ++j) nt[3+j] *= bs[j];
  }
  _Float16* np_ = nt_buf + ((size_t)n*HW512 + p)*32;
  #pragma unroll
  for (int c = 0; c < 16; ++c) np_[c] = (_Float16)nt[c];
  #pragma unroll
  for (int c = 16; c < 32; ++c) np_[c] = (_Float16)0.f;
}

// ---------------------------------------------------------------------------
// Weight pack: OIHW f32 -> per-lane WMMA B-fragment order, f16.
// Layout: [tap][kb][c16][lane(32)][j(16)] ; per ISA 16-bit B layout:
// lanes 0-15 hold N=lane, K 0..7 (j0..7) & 16..23 (j8..15);
// lanes 16-31 hold N=lane-16, K 8..15 & 24..31.
// ---------------------------------------------------------------------------
__global__ __launch_bounds__(256) void pack_weights_frag_kernel(
    const float* __restrict__ w, _Float16* __restrict__ wpf,
    int Cin, int Cout, int KB, int NC16)
{
  int i = blockIdx.x * blockDim.x + threadIdx.x;
  int total = 9 * KB * NC16 * 32 * 16;
  if (i >= total) return;
  int j = i & 15;
  int r = i >> 4;
  int lane = r & 31; r >>= 5;
  int c16 = r % NC16; r /= NC16;
  int kb = r % KB;
  int tap = r / KB;
  int half = lane >> 4;
  int k = (j < 8) ? (half*8 + j) : (16 + half*8 + (j - 8));
  int nn = lane & 15;
  int co = c16*16 + nn;
  int c = kb*32 + k;
  float val = (co < Cout && c < Cin) ? w[((size_t)co*Cin + c)*9 + tap] : 0.f;
  wpf[i] = (_Float16)val;
}

// ---------------------------------------------------------------------------
// 3x3 SAME conv, implicit GEMM via v_wmma_f32_16x16x32_f16.
// Activations NHWC f16, channels padded to multiple of 32.
// Block: 128 thr = 4 waves. M = 32 pixels (two 16-tiles, B reused for 2 WMMA),
// wave n owns 16 output channels. A staged to LDS with async b128 copies.
// Supports stride2, fused nearest up2, virtual concat (src0 ++ src1), lrelu.
// ---------------------------------------------------------------------------
__global__ __launch_bounds__(128) void conv3x3_wmma_kernel(
    const _Float16* __restrict__ src0, int kb0,
    const _Float16* __restrict__ src1, int kb1,
    const _Float16* __restrict__ wpf, const float* __restrict__ bias,
    _Float16* __restrict__ dst,
    int Cout, int CoutPad,
    int Hin, int Win, int HL, int WL, int Hout, int Wout,
    int stride, int up2, int pad_begin, int do_lrelu, int nCB)
{
  __shared__ __align__(16) _Float16 Atile[32][32];

  const int t    = threadIdx.x;
  const int wave = t >> 5;
  const int lane = t & 31;
  const int x0 = blockIdx.x * 32;
  const int y  = blockIdx.y;
  const int zb = blockIdx.z;
  const int n  = zb / nCB;
  const int co16 = (zb - n*nCB)*64 + wave*16;
  const bool active = (co16 < CoutPad);   // wave-uniform
  const int NC16 = CoutPad >> 4;
  const int c16  = co16 >> 4;
  const int KB = kb0 + kb1;
  const int Cs0 = kb0 << 5, Cs1 = kb1 << 5;

  // staging mapping: thread -> (pixel m, 16B channel chunk)
  const int m_st  = t >> 2;          // 0..31
  const int k0_st = (t & 3) << 3;    // half-index 0,8,16,24
  const unsigned lds_off = (unsigned)(size_t)&Atile[m_st][k0_st];

  v8f acc0 = {}, acc1 = {};

  for (int tap = 0; tap < 9; ++tap) {
    const int ty = tap / 3, tx = tap - ty*3;
    const int yl = y*stride + ty - pad_begin;
    const bool yok = (yl >= 0) && (yl < HL);
    const int yp = up2 ? (yl >> 1) : yl;
    const int xl = (x0 + m_st)*stride + tx - pad_begin;
    const bool ok0 = yok && (xl >= 0) && (xl < WL);
    const int xp = up2 ? (xl >> 1) : xl;

    for (int kb = 0; kb < KB; ++kb) {
      const _Float16* sp; int cb, Cs;
      if (kb < kb0) { sp = src0; cb = kb;       Cs = Cs0; }
      else          { sp = src1; cb = kb - kb0; Cs = Cs1; }

      __syncthreads();   // previous iteration's readers done
      if (ok0) {
        const _Float16* g =
            sp + ((size_t)(n*Hin + yp)*Win + xp)*Cs + (cb << 5) + k0_st;
        async_copy_b128(lds_off, g);     // 16B gather -> LDS, ASYNCcnt
      } else {
        v8h z = {};
        *(v8h*)&Atile[m_st][k0_st] = z;  // zero border
      }
      // B fragment straight from global (fragment-packed, coalesced b128 x2)
      v8h blo = {}, bhi = {};
      if (active) {
        const _Float16* bp =
            wpf + ((((size_t)tap*KB + kb)*NC16 + c16) << 9) + (lane << 4);
        blo = *(const v8h*)bp;
        bhi = *(const v8h*)(bp + 8);
      }
      async_wait0();
      __syncthreads();
      if (active) {
        v16h bf = __builtin_shufflevector(blo, bhi,
                    0,1,2,3,4,5,6,7,8,9,10,11,12,13,14,15);
        const int row  = lane & 15;
        const int hsel = (lane >> 4) << 3;
        v8h a0lo = *(const v8h*)&Atile[row][hsel];
        v8h a0hi = *(const v8h*)&Atile[row][hsel + 16];
        v8h a1lo = *(const v8h*)&Atile[16 + row][hsel];
        v8h a1hi = *(const v8h*)&Atile[16 + row][hsel + 16];
        v16h a0 = __builtin_shufflevector(a0lo, a0hi,
                    0,1,2,3,4,5,6,7,8,9,10,11,12,13,14,15);
        v16h a1 = __builtin_shufflevector(a1lo, a1hi,
                    0,1,2,3,4,5,6,7,8,9,10,11,12,13,14,15);
        acc0 = __builtin_amdgcn_wmma_f32_16x16x32_f16(
                   false, a0, false, bf, (short)0, acc0, false, false);
        acc1 = __builtin_amdgcn_wmma_f32_16x16x32_f16(
                   false, a1, false, bf, (short)0, acc1, false, false);
      }
    }
  }

  if (active) {
    const int co = co16 + (lane & 15);
    const float bv = (co < Cout) ? bias[co] : 0.f;
    #pragma unroll
    for (int mt = 0; mt < 2; ++mt) {
      v8f a = mt ? acc1 : acc0;
      #pragma unroll
      for (int r = 0; r < 8; ++r) {
        int m = mt*16 + r + ((lane >> 4) << 3);
        float vv = a[r] + bv;
        if (do_lrelu) vv = lrelu02(vv);
        dst[((size_t)(n*Hout + y)*Wout + (x0 + m))*CoutPad + co] = (_Float16)vv;
      }
    }
  }
}

// ---------------------------------------------------------------------------
// Final combine: final[n,c] = sum_k f[n,3k+c] * T[n,3k+c]  (T is NHWC/32)
// ---------------------------------------------------------------------------
__global__ __launch_bounds__(256) void combine_kernel(
    const float* __restrict__ f_buf, const _Float16* __restrict__ Tbuf,
    float* __restrict__ out_final)
{
  int gid = blockIdx.x * blockDim.x + threadIdx.x;
  if (gid >= 2 * 3 * HW512) return;
  int n = gid / (3 * HW512);
  int r = gid - n * (3 * HW512);
  int c = r / HW512;
  int p = r - c * HW512;
  const _Float16* tp = Tbuf + ((size_t)n*HW512 + p)*32;
  float s = 0.f;
  #pragma unroll
  for (int k = 0; k < 9; ++k) {
    s += f_buf[((size_t)n*27 + k*3 + c)*HW512 + p] * (float)tp[k*3 + c];
  }
  out_final[gid] = s;
}

__global__ __launch_bounds__(256) void copy_f32_kernel(
    const float* __restrict__ src, float* __restrict__ dst, int n)
{
  int i = blockIdx.x * blockDim.x + threadIdx.x;
  if (i < n) dst[i] = src[i];
}

// ---------------------------------------------------------------------------
extern "C" void kernel_launch(void* const* d_in, const int* in_sizes, int n_in,
                              void* d_out, int out_size, void* d_ws, size_t ws_size,
                              hipStream_t stream) {
  const float* wi        = (const float*)d_in[0];
  const float* envmap    = (const float*)d_in[1];
  const float* uv_map    = (const float*)d_in[2];
  const float* extr      = (const float*)d_in[3];
  const float* albedo_l  = (const float*)d_in[4];
  const float* tex_l1    = (const float*)d_in[5];
  const float* tex_l2    = (const float*)d_in[6];
  const float* tex_l3    = (const float*)d_in[7];
  const float* tex_l4    = (const float*)d_in[8];
  const float* w_d1 = (const float*)d_in[9];  const float* b_d1 = (const float*)d_in[10];
  const float* w_d2 = (const float*)d_in[11]; const float* b_d2 = (const float*)d_in[12];
  const float* w_d3 = (const float*)d_in[13]; const float* b_d3 = (const float*)d_in[14];
  const float* w_u3 = (const float*)d_in[15]; const float* b_u3 = (const float*)d_in[16];
  const float* w_u2 = (const float*)d_in[17]; const float* b_u2 = (const float*)d_in[18];
  const float* w_u1 = (const float*)d_in[19]; const float* b_u1 = (const float*)d_in[20];

  float* out = (float*)d_out;
  float* out_albedo_tex = out;                       // (1,3,512,512)
  float* out_final      = out + 786432;              // (2,3,512,512)
  float* out_fwd        = out + 786432 + 1572864;    // (2,3,512,512)
  float* out_albedo     = out + 786432 + 2*1572864;  // (2,3,512,512)

  // Workspace layout (activations NHWC f16, channels padded to x32)
  char* ws = (char*)d_ws;
  size_t off = 0;
  float*    f_buf = (float*)(ws + off);    off += (size_t)2*27*HW512*4;      // 56.6MB
  _Float16* ntb   = (_Float16*)(ws + off); off += (size_t)2*HW512*32*2;      // 33.6MB
  _Float16* d1b   = (_Float16*)(ws + off); off += (size_t)2*256*256*32*2;
  _Float16* d2b   = (_Float16*)(ws + off); off += (size_t)2*128*128*64*2;
  _Float16* d3b   = (_Float16*)(ws + off); off += (size_t)2*64*64*128*2;
  _Float16* u3b   = (_Float16*)(ws + off); off += (size_t)2*128*128*64*2;
  _Float16* u2b   = (_Float16*)(ws + off); off += (size_t)2*256*256*32*2;
  _Float16* Tb    = (_Float16*)(ws + off); off += (size_t)2*HW512*32*2;      // 33.6MB
  _Float16* wpd1  = (_Float16*)(ws + off); off += (size_t)9*1*2*512*2;
  _Float16* wpd2  = (_Float16*)(ws + off); off += (size_t)9*1*4*512*2;
  _Float16* wpd3  = (_Float16*)(ws + off); off += (size_t)9*2*8*512*2;
  _Float16* wpu3  = (_Float16*)(ws + off); off += (size_t)9*4*4*512*2;
  _Float16* wpu2  = (_Float16*)(ws + off); off += (size_t)9*4*2*512*2;
  _Float16* wpu1  = (_Float16*)(ws + off); off += (size_t)9*2*2*512*2;

  // Stage 1
  stage1_kernel<<<(2*HW512 + 255)/256, 256, 0, stream>>>(
      wi, envmap, uv_map, extr, albedo_l, tex_l1, tex_l2, tex_l3, tex_l4,
      out_fwd, out_albedo, f_buf, ntb);

  // albedo_tex passthrough
  copy_f32_kernel<<<(786432 + 255)/256, 256, 0, stream>>>(albedo_l, out_albedo_tex, 786432);

  // Pack weights into per-lane fragment order
  auto pack = [&](const float* w, _Float16* wp, int Cin, int Cout, int KB, int NC16) {
    int total = 9 * KB * NC16 * 32 * 16;
    pack_weights_frag_kernel<<<(total + 255)/256, 256, 0, stream>>>(w, wp, Cin, Cout, KB, NC16);
  };
  pack(w_d1, wpd1, 16,  32,  1, 2);
  pack(w_d2, wpd2, 32,  64,  1, 4);
  pack(w_d3, wpd3, 64,  128, 2, 8);
  pack(w_u3, wpu3, 128, 64,  4, 4);
  pack(w_u2, wpu2, 128, 32,  4, 2);
  pack(w_u1, wpu1, 64,  27,  2, 2);

  auto conv = [&](const _Float16* s0, int kb0, const _Float16* s1, int kb1,
                  const _Float16* wp, const float* bias, _Float16* dst,
                  int Cout, int CoutPad, int Hin, int Win, int HL, int WL,
                  int Hout, int Wout, int stride, int up2, int pad, int lrelu) {
    int nCB = (CoutPad + 63) / 64;
    dim3 grid(Wout / 32, Hout, 2 * nCB);
    conv3x3_wmma_kernel<<<grid, 128, 0, stream>>>(
        s0, kb0, s1, kb1, wp, bias, dst, Cout, CoutPad,
        Hin, Win, HL, WL, Hout, Wout, stride, up2, pad, lrelu, nCB);
  };

  // d1: nt(32p@512^2) -> 32 @ 256^2, stride2 (SAME pad_begin 0), lrelu
  conv(ntb, 1, nullptr, 0, wpd1, b_d1, d1b, 32, 32, 512, 512, 512, 512, 256, 256, 2, 0, 0, 1);
  // d2: 32 @ 256^2 -> 64 @ 128^2
  conv(d1b, 1, nullptr, 0, wpd2, b_d2, d2b, 64, 64, 256, 256, 256, 256, 128, 128, 2, 0, 0, 1);
  // d3: 64 @ 128^2 -> 128 @ 64^2
  conv(d2b, 2, nullptr, 0, wpd3, b_d3, d3b, 128, 128, 128, 128, 128, 128, 64, 64, 2, 0, 0, 1);
  // u3: up2(d3) 128 @ 128^2 -> 64 @ 128^2, stride1, pad1, lrelu
  conv(d3b, 4, nullptr, 0, wpu3, b_u3, u3b, 64, 64, 64, 64, 128, 128, 128, 128, 1, 1, 1, 1);
  // u2: up2(concat[u3, d2]) 128 @ 256^2 -> 32 @ 256^2
  conv(u3b, 2, d2b, 2, wpu2, b_u2, u2b, 32, 32, 128, 128, 256, 256, 256, 256, 1, 1, 1, 1);
  // u1: up2(concat[u2, d1]) 64 @ 512^2 -> 27 @ 512^2, no lrelu
  conv(u2b, 1, d1b, 1, wpu1, b_u1, Tb, 27, 32, 256, 256, 512, 512, 512, 512, 1, 1, 1, 0);

  // final = sum_k f .* T
  combine_kernel<<<(2*3*HW512 + 255)/256, 256, 0, stream>>>(f_buf, Tb, out_final);
}
